// BGD_67645734912743
// MI455X (gfx1250) — matrix operand order repaired and checked by reading, not compiled, not run on hardware
//
#include <hip/hip_runtime.h>
#include <hip/hip_bf16.h>

// ---------------------------------------------------------------------------
// MI455X (gfx1250) implementation of the corr-matrix AE + GCN pipeline.
// B=128, N=400 (pad 416), H=128.  All heavy GEMMs run on
// v_wmma_f32_16x16x32_bf16 with 32x32 register tiles per wave (4 WMMAs per
// K-step, 2x register reuse of each fragment) and fused
// bias/ReLU/residual/transposed stores.
// Top-k (k=32000/160000) per batch via 4-pass 8-bit radix select in LDS.
// ---------------------------------------------------------------------------

#define BB 128
#define NN 400
#define NP 416        // NN padded to multiple of 32 for WMMA K loops
#define HH 128
#define NTOT (NN * NN)      // 160000
#define KTOP 32000          // int(N*N*0.2)
#define EPSF 1e-9f

typedef unsigned short u16;
typedef __attribute__((ext_vector_type(16))) __bf16 v16bf;
typedef __attribute__((ext_vector_type(8)))  float  v8f;

union FragAB { uint4 q[2]; v16bf v; };

__device__ __forceinline__ u16 f2bf(float f) {
    unsigned u = __float_as_uint(f);
    unsigned r = (u + 0x7FFFu + ((u >> 16) & 1u)) >> 16;   // RNE
    return (u16)r;
}

__device__ __forceinline__ unsigned f2ord(float f) {
    unsigned u = __float_as_uint(f);
    return (u & 0x80000000u) ? ~u : (u | 0x80000000u);     // order-preserving
}

// ---------------------------------------------------------------------------
// Generic batched bf16 WMMA GEMM:  C[m,n] = sum_k A[m,k] * Bt[n,k] (+bias,res)
// A: row-major [M,K] bf16 (K contiguous).  Bt: "B transposed" [Nout,K] bf16.
// Wave tile: 32(M) x 32(N) = 2x2 WMMA tiles. 8 waves/block as 2(M)x4(N)
// -> 64(M) x 128(N) block tile.  K must be a multiple of 32 (pads zeroed).
// Partial edge subtiles: loads clamped to a valid row, stores suppressed.
// ---------------------------------------------------------------------------
template<bool HAS_BIAS, bool HAS_RES, bool RES_T, bool RELU,
         bool OUT_F32, bool OUT_BF, bool OUT_BFT>
__global__ __launch_bounds__(256) void gemm_bf16_wmma(
    const u16* __restrict__ A,  int lda,  long long sA,
    const u16* __restrict__ Bt, int ldb,  long long sB,
    const float* __restrict__ bias,
    const float* __restrict__ res, int ldres, long long sR,
    float* __restrict__ C,   int ldc,  long long sC,
    u16*   __restrict__ Cb,  int ldcb, long long sCb,
    u16*   __restrict__ CbT, int ldct, long long sCt,
    int M, int Nout, int K)
{
    const int lane = threadIdx.x & 31;
    const int wave = threadIdx.x >> 5;       // 0..7
    const int wm   = wave >> 2;              // 0..1
    const int wn   = wave & 3;               // 0..3
    const int m0 = blockIdx.y * 64  + wm * 32;
    const int n0 = blockIdx.x * 128 + wn * 32;
    const int b  = blockIdx.z;
    if (n0 >= Nout || m0 >= M) return;
    const bool vm1 = (m0 + 32) <= M;         // second M-subtile fully valid
    const bool vn1 = (n0 + 32) <= Nout;      // second N-subtile fully valid

    const int half = lane >> 4;     // 0: lanes 0-15, 1: lanes 16-31
    const int lr   = lane & 15;

    // A fragment: row = mbase+lr; half 0 holds K {0..7,16..23}, half 1 {8..15,24..31}
    const u16* Abase = A + (long long)b * sA + half * 8;
    const u16* Ap0 = Abase + (long long)(m0 + lr) * lda;
    const u16* Ap1 = vm1 ? (Abase + (long long)(m0 + 16 + lr) * lda) : Ap0;
    // B fragment from Bt: col n = nbase+lr; half 0 holds K {0..7,16..23}, half 1 {8..15,24..31}
    const u16* Bbase = Bt + (long long)b * sB + half * 8;
    const u16* Bp0 = Bbase + (long long)(n0 + lr) * ldb;
    const u16* Bp1 = vn1 ? (Bbase + (long long)(n0 + 16 + lr) * ldb) : Bp0;

    v8f acc00 = {0.f,0.f,0.f,0.f,0.f,0.f,0.f,0.f};
    v8f acc01 = acc00, acc10 = acc00, acc11 = acc00;

    for (int k0 = 0; k0 < K; k0 += 32) {
        FragAB fa0, fa1, fb0, fb1;
        fa0.q[0] = *(const uint4*)(Ap0 + k0);
        fa0.q[1] = *(const uint4*)(Ap0 + k0 + 16);
        fa1.q[0] = *(const uint4*)(Ap1 + k0);
        fa1.q[1] = *(const uint4*)(Ap1 + k0 + 16);
        fb0.q[0] = *(const uint4*)(Bp0 + k0);
        fb0.q[1] = *(const uint4*)(Bp0 + k0 + 16);
        fb1.q[0] = *(const uint4*)(Bp1 + k0);
        fb1.q[1] = *(const uint4*)(Bp1 + k0 + 16);
        if (k0 + 32 < K) {                       // global_prefetch_b8 next tile
            __builtin_prefetch(Ap0 + k0 + 32, 0, 1);
            __builtin_prefetch(Ap1 + k0 + 32, 0, 1);
            __builtin_prefetch(Bp0 + k0 + 32, 0, 1);
            __builtin_prefetch(Bp1 + k0 + 32, 0, 1);
        }
        acc00 = __builtin_amdgcn_wmma_f32_16x16x32_bf16(
                    false, fa0.v, false, fb0.v, (short)0, acc00, false, false);
        acc01 = __builtin_amdgcn_wmma_f32_16x16x32_bf16(
                    false, fa0.v, false, fb1.v, (short)0, acc01, false, false);
        acc10 = __builtin_amdgcn_wmma_f32_16x16x32_bf16(
                    false, fa1.v, false, fb0.v, (short)0, acc10, false, false);
        acc11 = __builtin_amdgcn_wmma_f32_16x16x32_bf16(
                    false, fa1.v, false, fb1.v, (short)0, acc11, false, false);
    }

    auto emit = [&](const v8f& acc, int mbase, int nbase) {
        const int n = nbase + lr;
        float bv = 0.0f;
        if (HAS_BIAS) bv = bias[n];
#pragma unroll
        for (int v = 0; v < 8; ++v) {
            const int m = mbase + half * 8 + v;  // C layout: VGPR v -> row v / v+8
            float val = acc[v] + bv;
            if (HAS_RES) {
                const float* rp = res + (long long)b * sR;
                val += RES_T ? rp[(long long)n * ldres + m]
                             : rp[(long long)m * ldres + n];
            }
            if (RELU) val = fmaxf(val, 0.0f);
            if (OUT_F32) C  [(long long)b * sC  + (long long)m * ldc  + n] = val;
            if (OUT_BF)  Cb [(long long)b * sCb + (long long)m * ldcb + n] = f2bf(val);
            if (OUT_BFT) CbT[(long long)b * sCt + (long long)n * ldct + m] = f2bf(val);
        }
    };
    emit(acc00, m0, n0);
    if (vn1)        emit(acc01, m0,      n0 + 16);
    if (vm1)        emit(acc10, m0 + 16, n0);
    if (vm1 && vn1) emit(acc11, m0 + 16, n0 + 16);
}

// ---------------------------------------------------------------------------
// Weight transpose + bf16 convert:  out[n, kp] = w[kp, n]  (kp >= K -> 0 pad)
// ---------------------------------------------------------------------------
__global__ void wT_bf16(const float* __restrict__ w, u16* __restrict__ out,
                        int K, int Nn, int Kpad)
{
    long long idx = (long long)blockIdx.x * 256 + threadIdx.x;
    long long tot = (long long)Nn * Kpad;
    if (idx >= tot) return;
    int n  = (int)(idx / Kpad);
    int kp = (int)(idx % Kpad);
    out[idx] = (kp < K) ? f2bf(w[(long long)kp * Nn + n]) : (u16)0;
}

// Zero the K-pad columns [NN, NP) of a [rows, NP] bf16 buffer.
__global__ void zero_pad16(u16* __restrict__ buf, int rows)
{
    int idx = blockIdx.x * 256 + threadIdx.x;
    if (idx >= rows * 16) return;
    int r = idx >> 4, c = NN + (idx & 15);
    buf[(long long)r * NP + c] = 0;
}

// ---------------------------------------------------------------------------
// Fisher transform + normalize by xf[b,0,0]; write f32 [B,N,N] + padded bf16.
// ---------------------------------------------------------------------------
__global__ void fisher_kernel(const float* __restrict__ x,
                              float* __restrict__ xf, u16* __restrict__ xfb)
{
    long long idx = (long long)blockIdx.x * 256 + threadIdx.x;
    const long long tot = (long long)BB * NN * NP;
    if (idx >= tot) return;
    int b   = (int)(idx / ((long long)NN * NP));
    int rem = (int)(idx % ((long long)NN * NP));
    int i = rem / NP, j = rem % NP;
    if (j < NN) {
        float x00 = x[(long long)b * NTOT];
        float f00 = 0.5f * logf((1.f + x00 + EPSF) / (1.f - x00 + EPSF));
        float v   = x[(long long)b * NTOT + (long long)i * NN + j];
        float f   = 0.5f * logf((1.f + v + EPSF) / (1.f - v + EPSF)) / f00;
        xf[(long long)b * NTOT + (long long)i * NN + j] = f;
        xfb[idx] = f2bf(f);
    } else {
        xfb[idx] = 0;
    }
}

// x2 = xf + 0.5*(delta + delta^T) off-diagonal;  write f32 + padded bf16.
__global__ void sym_kernel(const float* __restrict__ xf,
                           const float* __restrict__ delta,
                           float* __restrict__ x2, u16* __restrict__ x2b)
{
    long long idx = (long long)blockIdx.x * 256 + threadIdx.x;
    const long long tot = (long long)BB * NN * NP;
    if (idx >= tot) return;
    int b   = (int)(idx / ((long long)NN * NP));
    int rem = (int)(idx % ((long long)NN * NP));
    int i = rem / NP, j = rem % NP;
    if (j < NN) {
        const float* D = delta + (long long)b * NTOT;
        float d = (i == j) ? 0.0f : 0.5f * (D[(long long)i * NN + j] + D[(long long)j * NN + i]);
        float v = xf[(long long)b * NTOT + (long long)i * NN + j] + d;
        x2 [(long long)b * NTOT + (long long)i * NN + j] = v;
        x2b[idx] = f2bf(v);
    } else {
        x2b[idx] = 0;
    }
}

// ---------------------------------------------------------------------------
// Per-batch top-k threshold: 4-pass radix select (8 bits/pass) in LDS.
// thresh[b] = ordered-uint value of the k-th largest element.
// ---------------------------------------------------------------------------
__global__ __launch_bounds__(256) void topk_thresh(const float* __restrict__ x2,
                                                   unsigned* __restrict__ thresh)
{
    __shared__ unsigned hist[256];
    __shared__ unsigned sh_prefix, sh_k;
    const int b = blockIdx.x;
    const float* p = x2 + (long long)b * NTOT;

    if (threadIdx.x == 0) { sh_prefix = 0u; sh_k = KTOP; }
    __syncthreads();

    for (int pass = 0; pass < 4; ++pass) {
        const int shift = 24 - pass * 8;
        hist[threadIdx.x] = 0u;
        __syncthreads();
        const unsigned prefix = sh_prefix;
        const unsigned himask = (pass == 0) ? 0u : (0xFFFFFFFFu << (shift + 8));
        for (int i = threadIdx.x; i < NTOT; i += 256) {
            unsigned u = f2ord(p[i]);
            if ((u & himask) == (prefix & himask))
                atomicAdd(&hist[(u >> shift) & 255u], 1u);
        }
        __syncthreads();
        if (threadIdx.x == 0) {
            unsigned rem = sh_k, cum = 0u;
            for (int bin = 255; bin >= 0; --bin) {
                unsigned c = hist[bin];
                if (cum + c >= rem) {
                    sh_prefix = prefix | ((unsigned)bin << shift);
                    sh_k = rem - cum;
                    break;
                }
                cum += c;
            }
        }
        __syncthreads();
    }
    if (threadIdx.x == 0) thresh[b] = sh_prefix;
}

// ---------------------------------------------------------------------------
// GCN norm: deg[b,j] = sum_i A[i,j] (masked, self-loop fill) -> dinv = deg^-1/2
// ---------------------------------------------------------------------------
__global__ void deg_kernel(const float* __restrict__ x2,
                           const unsigned* __restrict__ thr,
                           float* __restrict__ dinv)
{
    int j = blockIdx.x * blockDim.x + threadIdx.x;
    int b = blockIdx.y;
    if (j >= NN) return;
    const float* p = x2 + (long long)b * NTOT;
    const unsigned t = thr[b];

    float dm = p[(long long)j * NN + j];
    dm = (f2ord(dm) >= t) ? dm : 0.0f;
    float deg = (dm == 0.0f) ? 1.0f : dm;      // add_remaining_self_loops fill=1
    for (int i = 0; i < NN; ++i) {
        if (i == j) continue;
        float a = p[(long long)i * NN + j];
        deg += (f2ord(a) >= t) ? a : 0.0f;
    }
    float dv = rsqrtf(deg);
    if (!(dv < 1e30f)) dv = 0.0f;              // inf/NaN -> 0
    dinv[b * NN + j] = dv;
}

// An^T bf16 operand (K-padded):  AnT[b, j, i] = dinv[i] * A[i,j] * dinv[j]
__global__ void ant_kernel(const float* __restrict__ x2,
                           const unsigned* __restrict__ thr,
                           const float* __restrict__ dinv,
                           u16* __restrict__ AnT)
{
    long long idx = (long long)blockIdx.x * 256 + threadIdx.x;
    const long long tot = (long long)BB * NN * NP;
    if (idx >= tot) return;
    int b   = (int)(idx / ((long long)NN * NP));
    int rem = (int)(idx % ((long long)NN * NP));
    int j = rem / NP, i = rem % NP;
    if (i >= NN) { AnT[idx] = 0; return; }
    const float* p = x2 + (long long)b * NTOT;
    const unsigned t = thr[b];
    float a = p[(long long)i * NN + j];
    a = (f2ord(a) >= t) ? a : 0.0f;
    if (i == j && a == 0.0f) a = 1.0f;         // self-loop fill
    AnT[idx] = f2bf(dinv[b * NN + i] * a * dinv[b * NN + j]);
}

// ---------------------------------------------------------------------------
// Head: pooled = mean_j h2[b,j,:];  out = relu(pooled) @ fc_w + fc_b
// ---------------------------------------------------------------------------
__global__ __launch_bounds__(128) void head_kernel(const float* __restrict__ h2,
                                                   const float* __restrict__ fc_w,
                                                   const float* __restrict__ fc_b,
                                                   float* __restrict__ out)
{
    __shared__ float acc0[HH], acc1[HH];
    const int b = blockIdx.x, f = threadIdx.x;
    const float* p = h2 + (long long)b * NN * HH;
    float s = 0.0f;
    for (int j = 0; j < NN; ++j) s += p[(long long)j * HH + f];
    float r = fmaxf(s / (float)NN, 0.0f);
    acc0[f] = r * fc_w[f * 2 + 0];
    acc1[f] = r * fc_w[f * 2 + 1];
    __syncthreads();
    for (int off = 64; off > 0; off >>= 1) {
        if (f < off) { acc0[f] += acc0[f + off]; acc1[f] += acc1[f + off]; }
        __syncthreads();
    }
    if (f == 0) {
        out[b * 2 + 0] = acc0[0] + fc_b[0];
        out[b * 2 + 1] = acc1[0] + fc_b[1];
    }
}

// ---------------------------------------------------------------------------
extern "C" void kernel_launch(void* const* d_in, const int* in_sizes, int n_in,
                              void* d_out, int out_size, void* d_ws, size_t ws_size,
                              hipStream_t stream)
{
    (void)in_sizes; (void)n_in; (void)out_size; (void)ws_size;

    const float* x      = (const float*)d_in[0];
    // d_in[1] = slength (unused by reference)
    const float* enc1_w = (const float*)d_in[2];
    const float* enc1_b = (const float*)d_in[3];
    const float* enc2_w = (const float*)d_in[4];
    const float* enc2_b = (const float*)d_in[5];
    const float* dec2_w = (const float*)d_in[6];
    const float* dec2_b = (const float*)d_in[7];
    const float* dec1_w = (const float*)d_in[8];
    const float* dec1_b = (const float*)d_in[9];
    const float* gcn1_w = (const float*)d_in[10];
    const float* gcn1_b = (const float*)d_in[11];
    const float* gcn2_w = (const float*)d_in[12];
    const float* gcn2_b = (const float*)d_in[13];
    const float* fc_w   = (const float*)d_in[14];
    const float* fc_b   = (const float*)d_in[15];
    float* out = (float*)d_out;

    // -------- workspace layout (256B aligned) --------
    char* ws = (char*)d_ws;
    size_t off = 0;
    auto alloc = [&](size_t bytes) { size_t o = off; off += (bytes + 255) & ~(size_t)255; return o; };

    float* xf     = (float*)(ws + alloc((size_t)BB * NTOT * 4));      // fisher f32
    u16*   xf_b   = (u16*)  (ws + alloc((size_t)BB * NN * NP * 2));   // fisher bf16 (padded)
    float* O1n    = (float*)(ws + alloc((size_t)BB * NN * HH * 4));   // out1 [N,H] f32 (residual)
    u16*   O1T    = (u16*)  (ws + alloc((size_t)BB * HH * NP * 2));   // out1^T [H,Np] bf16
    u16*   out2b  = (u16*)  (ws + alloc((size_t)BB * HH * HH * 2));   // out2 [H,H] bf16
    u16*   O2T    = (u16*)  (ws + alloc((size_t)BB * NN * HH * 2));   // out2' ^T [N,H] bf16
    float* delta  = (float*)(ws + alloc((size_t)BB * NTOT * 4));      // delta f32
    float* x2     = (float*)(ws + alloc((size_t)BB * NTOT * 4));      // x2 f32
    u16*   x2b    = (u16*)  (ws + alloc((size_t)BB * NN * NP * 2));   // x2 bf16 (padded)
    u16*   AnT    = (u16*)  (ws + alloc((size_t)BB * NN * NP * 2));   // An^T bf16 (padded)
    u16*   g1T    = (u16*)  (ws + alloc((size_t)BB * HH * NP * 2));   // (xp@W1)^T bf16
    u16*   hbuf   = (u16*)  (ws + alloc((size_t)BB * NN * HH * 2));   // h bf16 [N,H]
    u16*   g2T    = (u16*)  (ws + alloc((size_t)BB * HH * NP * 2));   // (h@W2)^T bf16
    float* h2     = (float*)(ws + alloc((size_t)BB * NN * HH * 4));   // h2 f32 [N,H]
    u16*   enc1wt = (u16*)  (ws + alloc((size_t)HH * NP * 2));
    u16*   enc2wt = (u16*)  (ws + alloc((size_t)HH * NP * 2));
    u16*   gcn1wt = (u16*)  (ws + alloc((size_t)HH * NP * 2));
    u16*   dec2wt = (u16*)  (ws + alloc((size_t)NN * HH * 2));
    u16*   dec1wt = (u16*)  (ws + alloc((size_t)NN * HH * 2));
    u16*   gcn2wt = (u16*)  (ws + alloc((size_t)HH * HH * 2));
    unsigned* thr = (unsigned*)(ws + alloc((size_t)BB * 4));
    float* dinv   = (float*)(ws + alloc((size_t)BB * NN * 4));

    const dim3 blk(256);

    // -------- weight transposes to bf16 (Bt layout, K-contiguous) --------
    {
        long long t1 = (long long)HH * NP, t2 = (long long)NN * HH, t3 = (long long)HH * HH;
        wT_bf16<<<dim3((unsigned)((t1 + 255) / 256)), blk, 0, stream>>>(enc1_w, enc1wt, NN, HH, NP);
        wT_bf16<<<dim3((unsigned)((t1 + 255) / 256)), blk, 0, stream>>>(enc2_w, enc2wt, NN, HH, NP);
        wT_bf16<<<dim3((unsigned)((t1 + 255) / 256)), blk, 0, stream>>>(gcn1_w, gcn1wt, NN, HH, NP);
        wT_bf16<<<dim3((unsigned)((t2 + 255) / 256)), blk, 0, stream>>>(dec2_w, dec2wt, HH, NN, HH);
        wT_bf16<<<dim3((unsigned)((t2 + 255) / 256)), blk, 0, stream>>>(dec1_w, dec1wt, HH, NN, HH);
        wT_bf16<<<dim3((unsigned)((t3 + 255) / 256)), blk, 0, stream>>>(gcn2_w, gcn2wt, HH, HH, HH);
    }
    // -------- zero the K-pad columns of transposed-activation buffers --------
    {
        int rows = BB * HH, tot = rows * 16;
        zero_pad16<<<dim3((tot + 255) / 256), blk, 0, stream>>>(O1T, rows);
        zero_pad16<<<dim3((tot + 255) / 256), blk, 0, stream>>>(g1T, rows);
        zero_pad16<<<dim3((tot + 255) / 256), blk, 0, stream>>>(g2T, rows);
    }

    const long long ewTot = (long long)BB * NN * NP;
    const dim3 ewGrid((unsigned)((ewTot + 255) / 256));

    // -------- fisher transform --------
    fisher_kernel<<<ewGrid, blk, 0, stream>>>(x, xf, xf_b);

    const long long sNNp = (long long)NN * NP;   // batched [N,Np] bf16 stride
    const long long sHNp = (long long)HH * NP;   // batched [H,Np] bf16 stride
    const long long sNH  = (long long)NN * HH;
    const long long sHH  = (long long)HH * HH;
    const long long sNNf = (long long)NTOT;

    const unsigned gyN = (NN + 63) / 64;   // M=400 -> 7 row-blocks
    const unsigned gyH = (HH + 63) / 64;   // M=128 -> 2 row-blocks

    // GEMM1: out1 = relu(xf @ enc1_w + b)   -> O1n f32 [N,H], O1T bf16 [H,Np]
    gemm_bf16_wmma<true, false, false, true, true, false, true>
        <<<dim3(1, gyN, BB), blk, 0, stream>>>(
            xf_b, NP, sNNp, enc1wt, NP, 0, enc1_b,
            nullptr, 0, 0, O1n, HH, sNH, nullptr, 0, 0, O1T, NP, sHNp,
            NN, HH, NP);

    // GEMM2: out2 = relu(out1^T-as-rows @ enc2_w + b) -> out2b bf16 [H,H]
    gemm_bf16_wmma<true, false, false, true, false, true, false>
        <<<dim3(1, gyH, BB), blk, 0, stream>>>(
            O1T, NP, sHNp, enc2wt, NP, 0, enc2_b,
            nullptr, 0, 0, nullptr, 0, 0, out2b, HH, sHH, nullptr, 0, 0,
            HH, HH, NP);

    // GEMM3: relu(out2 @ dec2_w + b + out1)  -> O2T bf16 [N,H] (transposed store)
    gemm_bf16_wmma<true, true, true, true, false, false, true>
        <<<dim3((NN + 127) / 128, gyH, BB), blk, 0, stream>>>(
            out2b, HH, sHH, dec2wt, HH, 0, dec2_b,
            O1n, HH, sNH, nullptr, 0, 0, nullptr, 0, 0, O2T, HH, sNH,
            HH, NN, HH);

    // GEMM4: delta = O2T @ dec1_w + b  -> delta f32 [N,N]
    gemm_bf16_wmma<true, false, false, false, true, false, false>
        <<<dim3((NN + 127) / 128, gyN, BB), blk, 0, stream>>>(
            O2T, HH, sNH, dec1wt, HH, 0, dec1_b,
            nullptr, 0, 0, delta, NN, sNNf, nullptr, 0, 0, nullptr, 0, 0,
            NN, NN, HH);

    // x2 = xf + 0.5*(delta + delta^T) off-diagonal
    sym_kernel<<<ewGrid, blk, 0, stream>>>(xf, delta, x2, x2b);

    // top-k threshold per batch (radix select)
    topk_thresh<<<dim3(BB), blk, 0, stream>>>(x2, thr);

    // degrees -> D^-1/2
    deg_kernel<<<dim3((NN + 255) / 256, BB), blk, 0, stream>>>(x2, thr, dinv);

    // An^T bf16 operand
    ant_kernel<<<ewGrid, blk, 0, stream>>>(x2, thr, dinv, AnT);

    // GEMM5: g1 = xp @ gcn1_w  -> g1T bf16 [H,Np] (transposed store)
    gemm_bf16_wmma<false, false, false, false, false, false, true>
        <<<dim3(1, gyN, BB), blk, 0, stream>>>(
            x2b, NP, sNNp, gcn1wt, NP, 0, nullptr,
            nullptr, 0, 0, nullptr, 0, 0, nullptr, 0, 0, g1T, NP, sHNp,
            NN, HH, NP);

    // GEMM6: h = relu(An^T @ g1 + b1)  -> hbuf bf16 [N,H]
    gemm_bf16_wmma<true, false, false, true, false, true, false>
        <<<dim3(1, gyN, BB), blk, 0, stream>>>(
            AnT, NP, sNNp, g1T, NP, sHNp, gcn1_b,
            nullptr, 0, 0, nullptr, 0, 0, hbuf, HH, sNH, nullptr, 0, 0,
            NN, HH, NP);

    // GEMM7: g2 = h @ gcn2_w  -> g2T bf16 [H,Np] (transposed store)
    gemm_bf16_wmma<false, false, false, false, false, false, true>
        <<<dim3(1, gyN, BB), blk, 0, stream>>>(
            hbuf, HH, sNH, gcn2wt, HH, 0, nullptr,
            nullptr, 0, 0, nullptr, 0, 0, nullptr, 0, 0, g2T, NP, sHNp,
            NN, HH, HH);

    // GEMM8: h2 = An^T @ g2 + b2  -> h2 f32 [N,H]
    gemm_bf16_wmma<true, false, false, false, true, false, false>
        <<<dim3(1, gyN, BB), blk, 0, stream>>>(
            AnT, NP, sNNp, g2T, NP, sHNp, gcn2_b,
            nullptr, 0, 0, h2, HH, sNH, nullptr, 0, 0, nullptr, 0, 0,
            NN, HH, NP);

    // head: mean-pool + relu + fc
    head_kernel<<<dim3(BB), dim3(HH), 0, stream>>>(h2, fc_w, fc_b, out);
}
